// GAT2_10428180595206
// MI455X (gfx1250) — matrix operand order repaired and checked by reading, not compiled, not run on hardware
//
#include <hip/hip_runtime.h>
#include <math.h>

// ---------------- problem constants (match reference) ----------------
#define NN     10000          // nodes
#define FIN    2613           // input features
#define KP1    2624           // FIN padded to multiple of 32
#define DD     256            // hidden dim
#define HH     4              // heads
#define HD     1024           // H*D
#define E0C    160000         // raw edges
#define ETOT   170000         // + self loops
#define BN_EPS 1e-5f
#define NEG_SLOPE 0.2f

typedef __attribute__((ext_vector_type(16))) __bf16 bf16x16;
typedef __attribute__((ext_vector_type(8)))  float  v8f;

struct u32x4 { unsigned int x, y, z, w; };

union Frag {
    bf16x16      v;
    unsigned int u[8];
};

// gfx1250 async LDS path (ASYNCcnt-tracked global->LDS DMA), guarded for portability
#if defined(__AMDGCN__) && __has_builtin(__builtin_amdgcn_global_load_async_to_lds_b128) && __has_builtin(__builtin_amdgcn_s_wait_asynccnt)
#define USE_ASYNC_LDS 1
typedef __attribute__((__vector_size__(4 * sizeof(int)))) int v4i;
typedef __attribute__((address_space(1))) v4i g_v4i;   // global-AS v4i
typedef __attribute__((address_space(3))) v4i l_v4i;   // LDS-AS v4i
#else
#define USE_ASYNC_LDS 0
#endif

// ---------------- helpers ----------------
__device__ __forceinline__ unsigned short f2bf(float f) {
    unsigned u = __float_as_uint(f);
    unsigned r = u + 0x7FFFu + ((u >> 16) & 1u);   // round-to-nearest-even
    return (unsigned short)(r >> 16);
}
__device__ __forceinline__ unsigned fenc(float f) {   // order-preserving uint key
    unsigned u = __float_as_uint(f);
    return (u & 0x80000000u) ? ~u : (u | 0x80000000u);
}
__device__ __forceinline__ float fdec(unsigned k) {
    return __uint_as_float((k & 0x80000000u) ? (k & 0x7FFFFFFFu) : ~k);
}
__device__ __forceinline__ void edge_sd(const long long* es, const long long* ed,
                                        int e, int& s, int& d) {
    if (e < E0C) {
        s = (int)es[e];
        int dd = (int)ed[e];
        d = (s == dd) ? ((dd + 1) % NN) : dd;     // reroute accidental self loops
    } else {
        s = d = e - E0C;                          // appended self loops
    }
}

// ---------------- bf16 cast / transpose kernels ----------------
__global__ void cast_pad_bf16(const float* __restrict__ src, unsigned short* __restrict__ dst,
                              int M, int K, int Kpad) {
    long long i = (long long)blockIdx.x * blockDim.x + threadIdx.x;
    long long tot = (long long)M * Kpad;
    if (i >= tot) return;
    int r = (int)(i / Kpad), c = (int)(i % Kpad);
    dst[i] = (c < K) ? f2bf(src[(long long)r * K + c]) : (unsigned short)0;
}

// W [K][N] fp32 -> Bt [N][Kpad] bf16 (zero padded)
__global__ void transpose_cast_bf16(const float* __restrict__ W, unsigned short* __restrict__ Bt,
                                    int K, int N, int Kpad) {
    long long i = (long long)blockIdx.x * blockDim.x + threadIdx.x;
    long long tot = (long long)N * Kpad;
    if (i >= tot) return;
    int n = (int)(i / Kpad), k = (int)(i % Kpad);
    Bt[i] = (k < K) ? f2bf(W[(long long)k * N + n]) : (unsigned short)0;
}

// ---------------- WMMA bf16 GEMM: C[M,N] = A[M,Kpad] * Bt[N,Kpad]^T ----------------
// block tile 128x128, 8 waves (2 in M x 4 in N), each wave: 4 Mfrag x 2 Nfrag.
// Double-buffered LDS; tile k+1 staged via GLOBAL_LOAD_ASYNC_TO_LDS_B128 while
// the 8 v_wmma_f32_16x16x32_bf16 of tile k execute. One barrier per K-step.
__device__ __forceinline__ void stage_tile(
    const unsigned short* __restrict__ A, const unsigned short* __restrict__ Bt,
    unsigned int (*As)[16], unsigned int (*Bs)[16],
    int M, int Kpad, int tileM, int tileN, int k0, int rA, int cg)
{
    for (int rb = 0; rb < 2; ++rb) {
        int row  = rA + rb * 64;
        int grow = tileM + row;
        if (grow >= M) grow = M - 1;   // clamp: tail rows compute garbage C rows we never store
        const unsigned short* ga = A  + (size_t)grow * Kpad + k0 + (cg << 1);
        const unsigned short* gb = Bt + (size_t)(tileN + row) * Kpad + k0 + (cg << 1);
#if USE_ASYNC_LDS
        __builtin_amdgcn_global_load_async_to_lds_b128(
            (g_v4i*)(size_t)ga, (l_v4i*)(v4i*)&As[row][cg], 0, 0);
        __builtin_amdgcn_global_load_async_to_lds_b128(
            (g_v4i*)(size_t)gb, (l_v4i*)(v4i*)&Bs[row][cg], 0, 0);
#else
        *(u32x4*)&As[row][cg] = *(const u32x4*)ga;
        *(u32x4*)&Bs[row][cg] = *(const u32x4*)gb;
#endif
    }
}

__global__ __launch_bounds__(256) void wmma_gemm_bf16(
    const unsigned short* __restrict__ A,   // bf16 [M][Kpad]
    const unsigned short* __restrict__ Bt,  // bf16 [N][Kpad]
    float* __restrict__ C,                  // fp32 [M][N]
    int M, int N, int Kpad)
{
    __shared__ unsigned int As[2][128][16];    // 2 x (128 rows x 32 bf16), 16 KB
    __shared__ unsigned int Bs[2][128][16];

    const int tid  = threadIdx.x;
    const int lane = tid & 31;
    const int wave = tid >> 5;
    const int wm   = wave >> 2;         // 0..1
    const int wn   = wave & 3;          // 0..3
    const int half = lane >> 4;         // 0/1
    const int l16  = lane & 15;
    const int tileM = blockIdx.y * 128;
    const int tileN = blockIdx.x * 128;

    v8f acc[4][2];
    for (int i = 0; i < 4; ++i)
        for (int j = 0; j < 2; ++j)
            for (int r = 0; r < 8; ++r) acc[i][j][r] = 0.0f;

    const int rA  = tid >> 2;           // 0..63
    const int cg  = (tid & 3) << 2;     // u32 group col: 0,4,8,12

    const int nk = Kpad >> 5;
    stage_tile(A, Bt, As[0], Bs[0], M, Kpad, tileM, tileN, 0, rA, cg);

    for (int ks = 0; ks < nk; ++ks) {
#if USE_ASYNC_LDS
        __builtin_amdgcn_s_wait_asynccnt(0);
#endif
        __syncthreads();                 // tile ks fully resident + prior reads of other buffer done
        const int cur = ks & 1;
        if (ks + 1 < nk)
            stage_tile(A, Bt, As[cur ^ 1], Bs[cur ^ 1], M, Kpad, tileM, tileN,
                       (ks + 1) << 5, rA, cg);

        // ---- build fragments per ISA 16-bit A/B layouts ----
        Frag af[4], bf[2];
        for (int i = 0; i < 4; ++i) {
            int row = wm * 64 + i * 16 + l16;
            *(u32x4*)&af[i].u[0] = *(const u32x4*)&As[cur][row][half * 4];       // K 0..7 / 8..15
            *(u32x4*)&af[i].u[4] = *(const u32x4*)&As[cur][row][8 + half * 4];   // K 16..23 / 24..31
        }
        for (int j = 0; j < 2; ++j) {
            int row = wn * 32 + j * 16 + l16;
            *(u32x4*)&bf[j].u[0] = *(const u32x4*)&Bs[cur][row][half * 4];
            *(u32x4*)&bf[j].u[4] = *(const u32x4*)&Bs[cur][row][8 + half * 4];
        }

        // ---- 8x v_wmma_f32_16x16x32_bf16 per K-step ----
        for (int i = 0; i < 4; ++i)
            for (int j = 0; j < 2; ++j)
                acc[i][j] = __builtin_amdgcn_wmma_f32_16x16x32_bf16(
                    false, af[i].v, false, bf[j].v, (short)0, acc[i][j], false, false);
        // no trailing barrier: next write targets the other LDS buffer
    }

    // ---- store C per 16x16 f32 C/D layout ----
    for (int i = 0; i < 4; ++i) {
        int rbase = tileM + wm * 64 + i * 16 + half * 8;
        for (int j = 0; j < 2; ++j) {
            int col = tileN + wn * 32 + j * 16 + l16;
            for (int r = 0; r < 8; ++r) {
                int row = rbase + r;
                if (row < M) C[(size_t)row * N + col] = acc[i][j][r];
            }
        }
    }
}

// ---------------- BatchNorm stats: one block per column ----------------
__global__ __launch_bounds__(256) void bn_stats(const float* __restrict__ Y, int M, int N,
                                                float* __restrict__ mean, float* __restrict__ rstd)
{
    __shared__ float ss[256], ss2[256];
    int c = blockIdx.x;
    float s = 0.f, s2 = 0.f;
    for (int r = threadIdx.x; r < M; r += 256) {
        float v = Y[(size_t)r * N + c];
        s += v; s2 += v * v;
    }
    ss[threadIdx.x] = s; ss2[threadIdx.x] = s2;
    __syncthreads();
    for (int o = 128; o > 0; o >>= 1) {
        if (threadIdx.x < o) { ss[threadIdx.x] += ss[threadIdx.x + o]; ss2[threadIdx.x] += ss2[threadIdx.x + o]; }
        __syncthreads();
    }
    if (threadIdx.x == 0) {
        float mu  = ss[0] / (float)M;
        float var = ss2[0] / (float)M - mu * mu;   // biased variance
        mean[c] = mu;
        rstd[c] = rsqrtf(var + BN_EPS);
    }
}

// ---------------- fused BN -> ELU -> bf16 cast ----------------
__global__ void bn_elu_cast(const float* __restrict__ Y,
                            const float* __restrict__ mean, const float* __restrict__ rstd,
                            const float* __restrict__ g, const float* __restrict__ b,
                            int M, int N, unsigned short* __restrict__ out)
{
    long long i = (long long)blockIdx.x * blockDim.x + threadIdx.x;
    long long tot = (long long)M * N;
    if (i >= tot) return;
    int c = (int)(i % N);
    float t = (Y[i] - mean[c]) * rstd[c] * g[c] + b[c];
    t = (t > 0.f) ? t : (expf(t) - 1.f);           // ELU, alpha=1
    out[i] = f2bf(t);
}

// ---------------- attention scores: one warp per (node, head) ----------------
__global__ void att_scores(const float* __restrict__ G,
                           const float* __restrict__ as, const float* __restrict__ ad,
                           float* __restrict__ As, float* __restrict__ Ad)
{
    int gid  = blockIdx.x * blockDim.x + threadIdx.x;
    int warp = gid >> 5, lane = gid & 31;
    if (warp >= NN * HH) return;
    int node = warp >> 2, h = warp & 3;
    const float* grow = G + (size_t)node * HD + h * DD;
    float s = 0.f, d = 0.f;
    for (int j = lane; j < DD; j += 32) {
        float gv = grow[j];
        s += gv * as[h * DD + j];
        d += gv * ad[h * DD + j];
    }
    for (int o = 16; o > 0; o >>= 1) {
        s += __shfl_down(s, o, 32);
        d += __shfl_down(d, o, 32);
    }
    if (lane == 0) { As[node * HH + h] = s; Ad[node * HH + h] = d; }
}

// ---------------- edge pass 1: leaky-relu logit + segment max ----------------
__global__ void edge_logits_max(const long long* __restrict__ es, const long long* __restrict__ ed,
                                const float* __restrict__ As, const float* __restrict__ Ad,
                                float* __restrict__ elr, unsigned int* __restrict__ mkey)
{
    int i = blockIdx.x * blockDim.x + threadIdx.x;
    if (i >= ETOT * HH) return;
    int e = i >> 2, h = i & 3;
    int s, d; edge_sd(es, ed, e, s, d);
    float v = As[s * HH + h] + Ad[d * HH + h];
    v = (v > 0.f) ? v : NEG_SLOPE * v;
    elr[i] = v;
    atomicMax(&mkey[d * HH + h], fenc(v));
}

// ---------------- edge pass 2: exp + segment sum ----------------
__global__ void edge_exp_sum(const long long* __restrict__ es, const long long* __restrict__ ed,
                             float* __restrict__ elr, const unsigned int* __restrict__ mkey,
                             float* __restrict__ den)
{
    int i = blockIdx.x * blockDim.x + threadIdx.x;
    if (i >= ETOT * HH) return;
    int e = i >> 2, h = i & 3;
    int s, d; edge_sd(es, ed, e, s, d);
    float w = expf(elr[i] - fdec(mkey[d * HH + h]));
    elr[i] = w;
    atomicAdd(&den[d * HH + h], w);
}

// ---------------- edge pass 3: weighted scatter-add (one block per edge) ----------------
__global__ __launch_bounds__(256) void edge_aggregate(
    const long long* __restrict__ es, const long long* __restrict__ ed,
    const float* __restrict__ elr, const float* __restrict__ den,
    const float* __restrict__ G, float* __restrict__ AGG)
{
    __shared__ float alpha[HH];
    int e = blockIdx.x;
    int s, d; edge_sd(es, ed, e, s, d);
    if (threadIdx.x < HH)
        alpha[threadIdx.x] = elr[e * HH + threadIdx.x] / (den[d * HH + threadIdx.x] + 1e-16f);
    __syncthreads();
    const float* gs = G + (size_t)s * HD;
    float* od = AGG + (size_t)d * HD;
    for (int j = threadIdx.x; j < HD; j += 256) {
        atomicAdd(&od[j], gs[j] * alpha[j >> 8]);
    }
}

// ---------------- finalize: + bg2 -> d_out ----------------
__global__ void finalize_bias(const float* __restrict__ AGG, const float* __restrict__ bias,
                              float* __restrict__ out)
{
    long long i = (long long)blockIdx.x * blockDim.x + threadIdx.x;
    long long tot = (long long)NN * HD;
    if (i >= tot) return;
    out[i] = AGG[i] + bias[(int)(i % HD)];
}

// ---------------- workspace layout (bytes, 256-aligned) ----------------
#define OFF_ABF   ((size_t)0)                               // bf16 activations, max 10000x2624
#define SZ_ABF    ((size_t)NN * KP1 * 2)                    // 52,480,000
#define OFF_WBF   (OFF_ABF + SZ_ABF)                        // bf16 weights [N][Kpad], max 1024x2624
#define SZ_WBF    ((size_t)1024 * KP1 * 2)                  // 5,373,952
#define OFF_G     (OFF_WBF + SZ_WBF)                        // fp32 GEMM out, max 10000x1024
#define SZ_G      ((size_t)NN * HD * 4)
#define OFF_AGG   (OFF_G + SZ_G)                            // fp32 aggregation out
#define SZ_AGG    ((size_t)NN * HD * 4)
#define OFF_ASRC  (OFF_AGG + SZ_AGG)
#define SZ_AH     ((size_t)NN * HH * 4)
#define OFF_ADST  (OFF_ASRC + SZ_AH)
#define OFF_MKEY  (OFF_ADST + SZ_AH)
#define OFF_DEN   (OFF_MKEY + SZ_AH)
#define OFF_ELR   (OFF_DEN + SZ_AH)
#define SZ_ELR    ((size_t)ETOT * HH * 4)
#define OFF_MEAN  (OFF_ELR + SZ_ELR)
#define OFF_RSTD  (OFF_MEAN + (size_t)4096)

static inline int ceil_div(long long a, int b) { return (int)((a + b - 1) / b); }

extern "C" void kernel_launch(void* const* d_in, const int* in_sizes, int n_in,
                              void* d_out, int out_size, void* d_ws, size_t ws_size,
                              hipStream_t stream)
{
    const float*     x    = (const float*)d_in[0];
    const long long* edges= (const long long*)d_in[1];
    const float*     W1   = (const float*)d_in[2];
    const float*     W2   = (const float*)d_in[4];
    const float*     W3   = (const float*)d_in[6];
    const float *g1=(const float*)d_in[8],  *be1=(const float*)d_in[9];
    const float *g2=(const float*)d_in[10], *be2=(const float*)d_in[11];
    const float *g3=(const float*)d_in[12], *be3=(const float*)d_in[13];
    const float *g4=(const float*)d_in[14], *be4=(const float*)d_in[15];
    const float* Wg1 = (const float*)d_in[16];
    const float *as1=(const float*)d_in[17], *ad1=(const float*)d_in[18];
    const float* Wg2 = (const float*)d_in[20];
    const float *as2=(const float*)d_in[21], *ad2=(const float*)d_in[22];
    const float* bg2 = (const float*)d_in[23];
    float* out = (float*)d_out;

    const long long* esrc = edges;
    const long long* edst = edges + E0C;

    char* ws = (char*)d_ws;
    unsigned short* ABF  = (unsigned short*)(ws + OFF_ABF);
    unsigned short* WBF  = (unsigned short*)(ws + OFF_WBF);
    float*          G    = (float*)(ws + OFF_G);
    float*          AGG  = (float*)(ws + OFF_AGG);
    float*          ASRC = (float*)(ws + OFF_ASRC);
    float*          ADST = (float*)(ws + OFF_ADST);
    unsigned int*   MKEY = (unsigned int*)(ws + OFF_MKEY);
    float*          DEN  = (float*)(ws + OFF_DEN);
    float*          ELR  = (float*)(ws + OFF_ELR);
    float*          MEAN = (float*)(ws + OFF_MEAN);
    float*          RSTD = (float*)(ws + OFF_RSTD);

    const int TB = 256;

    // ===== MLP layer 1: x @ W1 -> BN -> ELU (b1 absorbed by BN) =====
    cast_pad_bf16<<<ceil_div((long long)NN * KP1, TB), TB, 0, stream>>>(x, ABF, NN, FIN, KP1);
    transpose_cast_bf16<<<ceil_div((long long)DD * KP1, TB), TB, 0, stream>>>(W1, WBF, FIN, DD, KP1);
    {
        dim3 grid(DD / 128, ceil_div(NN, 128));
        wmma_gemm_bf16<<<grid, TB, 0, stream>>>(ABF, WBF, G, NN, DD, KP1);
    }
    bn_stats<<<DD, TB, 0, stream>>>(G, NN, DD, MEAN, RSTD);
    bn_elu_cast<<<ceil_div((long long)NN * DD, TB), TB, 0, stream>>>(G, MEAN, RSTD, g1, be1, NN, DD, ABF);

    // ===== MLP layers 2 & 3 =====
    const float* Ws[2]  = {W2, W3};
    const float* gs[2]  = {g2, g3};
    const float* bes[2] = {be2, be3};
    for (int l = 0; l < 2; ++l) {
        transpose_cast_bf16<<<ceil_div((long long)DD * DD, TB), TB, 0, stream>>>(Ws[l], WBF, DD, DD, DD);
        dim3 grid(DD / 128, ceil_div(NN, 128));
        wmma_gemm_bf16<<<grid, TB, 0, stream>>>(ABF, WBF, G, NN, DD, DD);
        bn_stats<<<DD, TB, 0, stream>>>(G, NN, DD, MEAN, RSTD);
        bn_elu_cast<<<ceil_div((long long)NN * DD, TB), TB, 0, stream>>>(G, MEAN, RSTD, gs[l], bes[l], NN, DD, ABF);
    }

    // ===== GAT layer 1: g = h @ Wg1 [256 -> 1024] =====
    transpose_cast_bf16<<<ceil_div((long long)HD * DD, TB), TB, 0, stream>>>(Wg1, WBF, DD, HD, DD);
    {
        dim3 grid(HD / 128, ceil_div(NN, 128));
        wmma_gemm_bf16<<<grid, TB, 0, stream>>>(ABF, WBF, G, NN, HD, DD);
    }
    att_scores<<<ceil_div((long long)NN * HH * 32, TB), TB, 0, stream>>>(G, as1, ad1, ASRC, ADST);
    (void)hipMemsetAsync(MKEY, 0, SZ_AH, stream);
    (void)hipMemsetAsync(DEN,  0, SZ_AH, stream);
    (void)hipMemsetAsync(AGG,  0, SZ_AGG, stream);
    edge_logits_max<<<ceil_div((long long)ETOT * HH, TB), TB, 0, stream>>>(esrc, edst, ASRC, ADST, ELR, MKEY);
    edge_exp_sum  <<<ceil_div((long long)ETOT * HH, TB), TB, 0, stream>>>(esrc, edst, ELR, MKEY, DEN);
    edge_aggregate<<<ETOT, TB, 0, stream>>>(esrc, edst, ELR, DEN, G, AGG);
    // bg1 absorbed by BN4
    bn_stats<<<HD, TB, 0, stream>>>(AGG, NN, HD, MEAN, RSTD);
    bn_elu_cast<<<ceil_div((long long)NN * HD, TB), TB, 0, stream>>>(AGG, MEAN, RSTD, g4, be4, NN, HD, ABF);

    // ===== GAT layer 2: g = h @ Wg2 [1024 -> 1024] =====
    transpose_cast_bf16<<<ceil_div((long long)HD * HD, TB), TB, 0, stream>>>(Wg2, WBF, HD, HD, HD);
    {
        dim3 grid(HD / 128, ceil_div(NN, 128));
        wmma_gemm_bf16<<<grid, TB, 0, stream>>>(ABF, WBF, G, NN, HD, HD);
    }
    att_scores<<<ceil_div((long long)NN * HH * 32, TB), TB, 0, stream>>>(G, as2, ad2, ASRC, ADST);
    (void)hipMemsetAsync(MKEY, 0, SZ_AH, stream);
    (void)hipMemsetAsync(DEN,  0, SZ_AH, stream);
    (void)hipMemsetAsync(AGG,  0, SZ_AGG, stream);
    edge_logits_max<<<ceil_div((long long)ETOT * HH, TB), TB, 0, stream>>>(esrc, edst, ASRC, ADST, ELR, MKEY);
    edge_exp_sum  <<<ceil_div((long long)ETOT * HH, TB), TB, 0, stream>>>(esrc, edst, ELR, MKEY, DEN);
    edge_aggregate<<<ETOT, TB, 0, stream>>>(esrc, edst, ELR, DEN, G, AGG);

    finalize_bias<<<ceil_div((long long)NN * HD, TB), TB, 0, stream>>>(AGG, bg2, out);
}
